// WeightOnlyInt8Linear_37778532336390
// MI455X (gfx1250) — compile-verified, matrix operand order checked by reading, not run on hardware
//
#include <hip/hip_runtime.h>
#include <stdint.h>

// ---- Problem shape (fixed by the reference) ----
static constexpr int M = 2 * 1024;   // B*S
static constexpr int N = 4096;
static constexpr int K = 4096;

// ---- Tiling ----
static constexpr int BM = 128;       // workgroup M tile
static constexpr int BN = 256;       // workgroup N tile
static constexpr int BK = 32;        // one WMMA K-step per LDS stage
static constexpr int LDP = BK + 8;   // padded LDS row stride (bf16), 80B rows, 16B aligned

typedef __attribute__((ext_vector_type(16))) __bf16 bf16x16;
typedef __attribute__((ext_vector_type(4)))  __bf16 bf16x4;
typedef __attribute__((ext_vector_type(8)))  float  f32x8;

// ---------------------------------------------------------------------------
// Stage 1a: x f32 -> bf16 (4 elements / thread)
// ---------------------------------------------------------------------------
__global__ __launch_bounds__(256) void cvt_f32_to_bf16(const float* __restrict__ in,
                                                       __bf16* __restrict__ out) {
  size_t i = ((size_t)blockIdx.x * 256 + threadIdx.x) * 4;
  float4 f = *(const float4*)(in + i);
  bf16x4 o;
  o[0] = (__bf16)f.x; o[1] = (__bf16)f.y; o[2] = (__bf16)f.z; o[3] = (__bf16)f.w;
  *(bf16x4*)(out + i) = o;
}

// ---------------------------------------------------------------------------
// Stage 1b: weight int -> bf16 (exact: |w| <= 128 fits bf16 significand)
// ---------------------------------------------------------------------------
__global__ __launch_bounds__(256) void cvt_i32_to_bf16(const int* __restrict__ in,
                                                       __bf16* __restrict__ out) {
  size_t i = ((size_t)blockIdx.x * 256 + threadIdx.x) * 4;
  int4 w = *(const int4*)(in + i);
  bf16x4 o;
  o[0] = (__bf16)(float)w.x; o[1] = (__bf16)(float)w.y;
  o[2] = (__bf16)(float)w.z; o[3] = (__bf16)(float)w.w;
  *(bf16x4*)(out + i) = o;
}

// ---------------------------------------------------------------------------
// Async global->LDS copy of 32 contiguous bytes (2 x b128), ASYNCcnt-tracked.
// INST_OFFSET applies to both global and LDS address (ISA §8 pseudocode).
// ---------------------------------------------------------------------------
__device__ __forceinline__ void async_copy32B(unsigned lds_addr, const __bf16* gptr) {
  asm volatile(
      "global_load_async_to_lds_b128 %0, %1, off\n\t"
      "global_load_async_to_lds_b128 %0, %1, off offset:16"
      :: "v"(lds_addr), "v"(gptr) : "memory");
}

// ---------------------------------------------------------------------------
// Stage 2: C = A(MxK) * W(NxK)^T via V_WMMA_F32_16X16X32_BF16.
// 128x256 tile, 8 waves in 2(M) x 4(N); each wave 64x64 -> 4x4 accumulators.
// Double-buffered LDS fed by async copies; K-loop unrolled x2 so buffer
// indices are compile-time (keeps WMMA accumulation in-place, no mov churn).
// ---------------------------------------------------------------------------
__global__ __launch_bounds__(256) void gemm_bf16_wmma(const __bf16* __restrict__ A,
                                                      const __bf16* __restrict__ W,
                                                      const float* __restrict__ scales,
                                                      const float* __restrict__ bias,
                                                      float* __restrict__ out) {
  __shared__ __bf16 sA[2][BM * LDP];   // 2 x 10 KB
  __shared__ __bf16 sB[2][BN * LDP];   // 2 x 20 KB

  const int tid  = threadIdx.x;
  const int m0   = blockIdx.y * BM;
  const int n0   = blockIdx.x * BN;

  const int wave = tid >> 5;
  const int lane = tid & 31;
  const int half = lane >> 4;   // K-subgroup select per ISA 16-bit A/B layout
  const int r    = lane & 15;

  const int wm = (wave & 1) * 64;   // wave M offset in tile
  const int wn = (wave >> 1) * 64;  // wave N offset in tile

  // Staging: 3 x 32B segments per thread.
  //   seg0 -> A row (tid>>1), seg1 -> B row (tid>>1), seg2 -> B row (tid>>1)+128
  const int srow    = tid >> 1;          // 0..127
  const int colBase = (tid & 1) * 16;    // 0 or 16

  const __bf16* gA  = A + (size_t)(m0 + srow) * K + colBase;
  const __bf16* gB0 = W + (size_t)(n0 + srow) * K + colBase;
  const __bf16* gB1 = W + (size_t)(n0 + srow + 128) * K + colBase;

  // LDS byte addresses (generic ptr low 32 bits == LDS offset).
  const unsigned ldsA[2] = {
      (unsigned)(uintptr_t)&sA[0][srow * LDP + colBase],
      (unsigned)(uintptr_t)&sA[1][srow * LDP + colBase]};
  const unsigned ldsB0[2] = {
      (unsigned)(uintptr_t)&sB[0][srow * LDP + colBase],
      (unsigned)(uintptr_t)&sB[1][srow * LDP + colBase]};
  const unsigned ldsB1[2] = {
      (unsigned)(uintptr_t)&sB[0][(srow + 128) * LDP + colBase],
      (unsigned)(uintptr_t)&sB[1][(srow + 128) * LDP + colBase]};

  f32x8 acc[4][4] = {};
  union Frag { bf16x16 v; uint4 u[2]; };

  auto stage = [&](int buf, int k0) {
    async_copy32B(ldsA[buf],  gA  + k0);
    async_copy32B(ldsB0[buf], gB0 + k0);
    async_copy32B(ldsB1[buf], gB1 + k0);
  };

  auto compute = [&](int buf) {
    Frag fa[4];
#pragma unroll
    for (int ii = 0; ii < 4; ++ii) {
      const int ar = wm + ii * 16 + r;
      fa[ii].u[0] = *(const uint4*)&sA[buf][ar * LDP + half * 8];
      fa[ii].u[1] = *(const uint4*)&sA[buf][ar * LDP + 16 + half * 8];
    }
    Frag fb[4];
#pragma unroll
    for (int j = 0; j < 4; ++j) {
      const int br = wn + j * 16 + r;
      fb[j].u[0] = *(const uint4*)&sB[buf][br * LDP + half * 8];
      fb[j].u[1] = *(const uint4*)&sB[buf][br * LDP + 16 + half * 8];
    }
#pragma unroll
    for (int ii = 0; ii < 4; ++ii)
#pragma unroll
      for (int j = 0; j < 4; ++j)
        acc[ii][j] = __builtin_amdgcn_wmma_f32_16x16x32_bf16(
            /*neg_a=*/false, fa[ii].v, /*neg_b=*/false, fb[j].v,
            /*c_mod=*/(short)0, acc[ii][j], /*reuse_a=*/false, /*reuse_b=*/false);
  };

  // Prologue: stage buffer 0 (k=0). 6 async loads in flight.
  stage(0, 0);

  constexpr int KSTEPS = K / BK;  // 128 (even)
  for (int i = 0; i < KSTEPS; i += 2) {
    // ---- even step: consume buffer 0 ----
    {
      // i+1 < KSTEPS always holds (KSTEPS even): stage buffer 1.
      stage(1, (i + 1) * BK);
      asm volatile("s_wait_asynccnt 0x6" ::: "memory");  // buffer-0 loads landed
      __syncthreads();
      compute(0);
      __syncthreads();  // buffer 0 retired
    }
    // ---- odd step: consume buffer 1 ----
    {
      if (i + 2 < KSTEPS) {
        stage(0, (i + 2) * BK);
        asm volatile("s_wait_asynccnt 0x6" ::: "memory");  // buffer-1 loads landed
      } else {
        asm volatile("s_wait_asynccnt 0x0" ::: "memory");
      }
      __syncthreads();
      compute(1);
      __syncthreads();  // buffer 1 retired
    }
  }

  // Epilogue: per-channel scale + bias.
  // C/D layout: VGPR p -> M = p + 8*half; lane r -> N = r (16x16 f32 table).
#pragma unroll
  for (int j = 0; j < 4; ++j) {
    const int n = n0 + wn + j * 16 + r;
    const float s = scales[n];
    const float bi = bias[n];
#pragma unroll
    for (int ii = 0; ii < 4; ++ii) {
#pragma unroll
      for (int p = 0; p < 8; ++p) {
        const int m = m0 + wm + ii * 16 + half * 8 + p;
        out[(size_t)m * N + n] = acc[ii][j][p] * s + bi;
      }
    }
  }
}

// ---------------------------------------------------------------------------
extern "C" void kernel_launch(void* const* d_in, const int* in_sizes, int n_in,
                              void* d_out, int out_size, void* d_ws, size_t ws_size,
                              hipStream_t stream) {
  const float* x      = (const float*)d_in[0];   // [2,1024,4096] f32
  const int*   weight = (const int*)d_in[1];     // [4096,4096] int
  const float* scales = (const float*)d_in[2];   // [4096] f32
  const float* bias   = (const float*)d_in[3];   // [4096] f32
  float* out = (float*)d_out;                    // [2,1024,4096] f32

  __bf16* Abf = (__bf16*)d_ws;                               // 16 MB
  __bf16* Wbf = (__bf16*)((char*)d_ws + (size_t)M * K * 2);  // +32 MB

  cvt_f32_to_bf16<<<(size_t)M * K / 1024, 256, 0, stream>>>(x, Abf);
  cvt_i32_to_bf16<<<(size_t)N * K / 1024, 256, 0, stream>>>(weight, Wbf);

  dim3 grid(N / BN, M / BM);  // 16 x 16 workgroups
  gemm_bf16_wmma<<<grid, 256, 0, stream>>>(Abf, Wbf, scales, bias, out);
}
